// TextLSTM1_85237920957101
// MI455X (gfx1250) — matrix operand order repaired and checked by reading, not compiled, not run on hardware
//
#include <hip/hip_runtime.h>
#include <hip/hip_bf16.h>

// ---------------------------------------------------------------------------
// TextLSTM on MI455X (gfx1250).
//   * bf16 WMMA (v_wmma_f32_16x16x32_bf16, f32 accum) for every GEMM
//   * weights transposed/converted to bf16 once per call (L2-resident, ~40MB)
//   * ONE persistent kernel runs the whole T=256 scan: 32 co-resident blocks,
//     device-scope atomic grid barriers between the 3 phases of each step.
//   * staging uses gfx1250 async global->LDS copies (ASYNCcnt) with
//     double-buffered LDS slots; falls back to sync copies if the builtin
//     is unavailable.
//   * final 256x32000 projection as a separate wide WMMA kernel
// ---------------------------------------------------------------------------

typedef __attribute__((ext_vector_type(16))) __bf16 bf16x16;
typedef __attribute__((ext_vector_type(8)))  float  f32x8;
typedef __attribute__((ext_vector_type(4)))  int    i32x4;

#define B_SZ   256
#define T_SZ   256
#define EMBD   256
#define HID    512
#define NCLASS 32000
#define KTOT   (EMBD + HID)   // 768
#define LDP    40             // padded LDS row stride (elems): 80B = 20 banks

#if defined(__has_builtin)
#  if __has_builtin(__builtin_amdgcn_global_load_async_to_lds_b128) && \
      __has_builtin(__builtin_amdgcn_s_wait_asynccnt)
#    define HAVE_ASYNC_LDS 1
#  endif
#endif
#ifndef HAVE_ASYNC_LDS
#  define HAVE_ASYNC_LDS 0
#endif

struct U32x8 { uint4 lo, hi; };

// 16-byte global -> LDS copy (async on gfx1250: global_load_async_to_lds_b128)
// Builtin signature (from compiler diagnostic): (int4 AS1*, int4 AS3*, Ii, Ii)
static __device__ inline void cp16(__bf16* lds, const __bf16* g) {
#if HAVE_ASYNC_LDS
  __builtin_amdgcn_global_load_async_to_lds_b128(
      (__attribute__((address_space(1))) i32x4*)(__bf16*)g,
      (__attribute__((address_space(3))) i32x4*)lds, 0, 0);
#else
  *(uint4*)lds = *(const uint4*)g;
#endif
}

// Wait until at most N of my async copies are still in flight.
template <int N> static __device__ inline void stage_wait() {
#if HAVE_ASYNC_LDS
  __builtin_amdgcn_s_wait_asynccnt(N);
#endif
}

// Build a per-lane WMMA fragment from two contiguous 16B LDS reads.
static __device__ inline bf16x16 frag_ld(const __bf16* p, int hioff_elems) {
  U32x8 u;
  u.lo = *(const uint4*)(p);
  u.hi = *(const uint4*)(p + hioff_elems);
  return __builtin_bit_cast(bf16x16, u);
}

static __device__ inline float sigf(float x) { return 1.0f / (1.0f + __expf(-x)); }

// Device-wide barrier: release fence + arrive, spin until all blocks arrived,
// acquire fence.  Monotonic counter; `target` = nblocks * barrier_index.
static __device__ inline void grid_barrier(unsigned* ctr, unsigned target) {
  __threadfence();                      // release my stores to device scope
  __syncthreads();
  if (threadIdx.x == 0) {
    __hip_atomic_fetch_add(ctr, 1u, __ATOMIC_ACQ_REL, __HIP_MEMORY_SCOPE_AGENT);
    while (__hip_atomic_load(ctr, __ATOMIC_ACQUIRE, __HIP_MEMORY_SCOPE_AGENT) < target) {
      __builtin_amdgcn_s_sleep(1);
    }
  }
  __syncthreads();
  __threadfence();                      // acquire: invalidate stale WGP$ lines
}

// --------------------------- prep kernels ----------------------------------

// dst[n][k] = (k<KA ? A[k][n] : Bsrc[k-KA][n]) as bf16.  A:[KA][N], Bsrc:[KB][N]
__global__ void k_cat_tr(const float* __restrict__ A, int KA,
                         const float* __restrict__ Bsrc, int KB,
                         int N, __bf16* __restrict__ dst) {
  const int K = KA + KB;
  const long long total = (long long)N * K;
  for (long long i = (long long)blockIdx.x * blockDim.x + threadIdx.x; i < total;
       i += (long long)gridDim.x * blockDim.x) {
    const int k = (int)(i % K);
    const int n = (int)(i / K);
    const float v = (k < KA) ? A[(size_t)k * N + n] : Bsrc[(size_t)(k - KA) * N + n];
    dst[i] = (__bf16)v;
  }
}

// pack 4 gate biases into one [4][HID] buffer (kills branchy epilogue loads)
__global__ void k_pack_bias(const float* __restrict__ b0, const float* __restrict__ b1,
                            const float* __restrict__ b2, const float* __restrict__ b3,
                            float* __restrict__ dst) {
  const int i = blockIdx.x * blockDim.x + threadIdx.x;
  if (i < 4 * HID) {
    const float* src = (i < HID) ? b0 : (i < 2 * HID) ? b1 : (i < 3 * HID) ? b2 : b3;
    dst[i] = src[i & (HID - 1)];
  }
}

// Xall[t][b][e] = bf16(E[X[b][t]][e])
__global__ void k_gather(const int* __restrict__ X, const float* __restrict__ E,
                         __bf16* __restrict__ Xall) {
  const long long total = (long long)T_SZ * B_SZ * EMBD;
  for (long long i = (long long)blockIdx.x * blockDim.x + threadIdx.x; i < total;
       i += (long long)gridDim.x * blockDim.x) {
    const int e = (int)(i % EMBD);
    const long long tb = i / EMBD;
    const int b  = (int)(tb % B_SZ);
    const int tt = (int)(tb / B_SZ);
    const int tok = X[(size_t)b * T_SZ + tt];
    Xall[i] = (__bf16)E[(size_t)tok * EMBD + e];
  }
}

__global__ void k_init(const float* __restrict__ H, const float* __restrict__ C,
                       __bf16* __restrict__ Hbf, __bf16* __restrict__ H1bf,
                       float* __restrict__ Cw, float* __restrict__ C1w,
                       unsigned* __restrict__ ctr) {
  if (blockIdx.x == 0 && threadIdx.x == 0) *ctr = 0u;   // reset grid barrier
  const int n = B_SZ * HID;
  for (int i = blockIdx.x * blockDim.x + threadIdx.x; i < n;
       i += gridDim.x * blockDim.x) {
    const float h = H[i];
    Hbf[i]  = (__bf16)h;
    H1bf[i] = (__bf16)h;
    const float c = C[i];
    Cw[i]  = c;
    C1w[i] = c;
  }
}

// --------------------------- phase bodies ----------------------------------
// One block owns a 64(batch) x 64(hid) tile for the whole scan.

// 4-gate LSTM cell tile: A = [Xt | Hin] (K=768), Wt[4][HID][KTOT], fused
// C = F*C + I*Ct ; Hout = O*tanh(C).  Double-buffered async LDS staging.
static __device__ inline void cell_tile(
    const __bf16* Xt, const __bf16* Hin, const __bf16* Wt,
    const float* biasAll /*[4][HID]*/, float* Cst, __bf16* Hout,
    int mb, int nb,
    __bf16* As /*[2][64*LDP]*/, __bf16* Bs /*[2][4*64*LDP]*/, float* Gact) {
  const int t = threadIdx.x;
  const int wave = t >> 5, lane = t & 31;
  const int gate = wave >> 1;          // 2 waves per gate
  const int mhalf = (wave & 1) * 32;   // each wave: 32 rows x 64 cols
  const int halfsel = lane >> 4;
  const int lm = lane & 15;
  const int arow = t >> 2, acol = (t & 3) * 8;   // A stage: 64 rows x 32
  const int bgate = t >> 6, brow = t & 63;       // B stage: 4 x 64 rows x 32
  constexpr int NC = KTOT / 32;                  // 24 chunks
  constexpr int ASLOT = 64 * LDP;
  constexpr int BSLOT = 4 * 64 * LDP;

  f32x8 acc[2][4] = {};

  auto issue = [&](int kc, int s) {
    const int ks = kc * 32;
    const __bf16* asrc = (ks < EMBD)
        ? (Xt  + (size_t)(mb + arow) * EMBD + ks)
        : (Hin + (size_t)(mb + arow) * HID + (ks - EMBD));
    cp16(As + s * ASLOT + arow * LDP + acol, asrc + acol);
    const __bf16* wsrc = Wt + ((size_t)bgate * HID + (nb + brow)) * KTOT + ks;
    __bf16* bd = Bs + s * BSLOT + (bgate * 64 + brow) * LDP;
    cp16(bd + 0,  wsrc + 0);
    cp16(bd + 8,  wsrc + 8);
    cp16(bd + 16, wsrc + 16);
    cp16(bd + 24, wsrc + 24);
  };                                   // 5 async copies per thread per chunk

  issue(0, 0);
  for (int kc = 0; kc < NC; ++kc) {
    const int s = kc & 1;
    if (kc + 1 < NC) { issue(kc + 1, s ^ 1); stage_wait<5>(); }
    else             { stage_wait<0>(); }
    __syncthreads();                   // chunk kc staged by all waves

    const __bf16* Asl = As + s * ASLOT;
    const __bf16* Bsl = Bs + s * BSLOT;
    bf16x16 af[2], bfr[4];
#pragma unroll
    for (int mt = 0; mt < 2; ++mt)
      af[mt] = frag_ld(&Asl[(mhalf + mt * 16 + lm) * LDP + halfsel * 8], 16);
#pragma unroll
    for (int nt = 0; nt < 4; ++nt)
      bfr[nt] = frag_ld(&Bsl[(gate * 64 + nt * 16 + lm) * LDP + halfsel * 8], 16);
#pragma unroll
    for (int mt = 0; mt < 2; ++mt)
#pragma unroll
      for (int nt = 0; nt < 4; ++nt)
        acc[mt][nt] = __builtin_amdgcn_wmma_f32_16x16x32_bf16(
            false, af[mt], false, bfr[nt], (short)0, acc[mt][nt], false, false);
    __syncthreads();                   // everyone done reading slot s
  }

  // bias + activation -> Gact (gate 3 = candidate uses tanh, others sigmoid)
  const bool isC = (gate == 3);
#pragma unroll
  for (int nt = 0; nt < 4; ++nt) {
    const int nl = nt * 16 + lm;
    const float bv = biasAll[gate * HID + nb + nl];
#pragma unroll
    for (int mt = 0; mt < 2; ++mt) {
#pragma unroll
      for (int j = 0; j < 8; ++j) {
        const int ml = mhalf + mt * 16 + j + 8 * halfsel;
        const float v = acc[mt][nt][j] + bv;
        Gact[((size_t)gate * 64 + ml) * 64 + nl] = isC ? tanhf(v) : sigf(v);
      }
    }
  }
  __syncthreads();

  // fused LSTM state update
  const int crow = t >> 2, ccol0 = (t & 3) * 16;
#pragma unroll
  for (int q = 0; q < 16; ++q) {
    const int col = ccol0 + q;
    const float Iv = Gact[(0 * 64 + crow) * 64 + col];
    const float Fv = Gact[(1 * 64 + crow) * 64 + col];
    const float Ov = Gact[(2 * 64 + crow) * 64 + col];
    const float Cv = Gact[(3 * 64 + crow) * 64 + col];
    const size_t gi = (size_t)(mb + crow) * HID + nb + col;
    const float c = Fv * Cst[gi] + Iv * Cv;
    Cst[gi] = c;
    Hout[gi] = (__bf16)(Ov * tanhf(c));
  }
}

// H1mix = sigmoid([Hn | H1p] @ Wt^T + bh), K = 1024
static __device__ inline void mix_tile(
    const __bf16* Hn, const __bf16* H1p, const __bf16* Wt /*[HID][2*HID]*/,
    const float* bh, __bf16* H1mix, int mb, int nb,
    __bf16* As /*[2][64*LDP]*/, __bf16* Bs /*[2][64*LDP]*/) {
  const int t = threadIdx.x;
  const int wave = t >> 5, lane = t & 31;
  const int mq = (wave & 3) * 16;      // wave: 16 rows x 32 cols
  const int nh = (wave >> 2) * 32;
  const int halfsel = lane >> 4, lm = lane & 15;
  const int arow = t >> 2, acol = (t & 3) * 8;
  constexpr int NC = (2 * HID) / 32;   // 32 chunks
  constexpr int SLOT = 64 * LDP;

  f32x8 acc[2] = {};

  auto issue = [&](int kc, int s) {
    const int ks = kc * 32;
    const __bf16* asrc = (ks < HID)
        ? (Hn  + (size_t)(mb + arow) * HID + ks)
        : (H1p + (size_t)(mb + arow) * HID + (ks - HID));
    cp16(As + s * SLOT + arow * LDP + acol, asrc + acol);
    const __bf16* wsrc = Wt + (size_t)(nb + arow) * (2 * HID) + ks;
    cp16(Bs + s * SLOT + arow * LDP + acol, wsrc + acol);
  };                                   // 2 async copies per thread per chunk

  issue(0, 0);
  for (int kc = 0; kc < NC; ++kc) {
    const int s = kc & 1;
    if (kc + 1 < NC) { issue(kc + 1, s ^ 1); stage_wait<2>(); }
    else             { stage_wait<0>(); }
    __syncthreads();
    const __bf16* Asl = As + s * SLOT;
    const __bf16* Bsl = Bs + s * SLOT;
    const bf16x16 af = frag_ld(&Asl[(mq + lm) * LDP + halfsel * 8], 16);
#pragma unroll
    for (int nt = 0; nt < 2; ++nt) {
      const bf16x16 bfr = frag_ld(&Bsl[(nh + nt * 16 + lm) * LDP + halfsel * 8], 16);
      acc[nt] = __builtin_amdgcn_wmma_f32_16x16x32_bf16(
          false, af, false, bfr, (short)0, acc[nt], false, false);
    }
    __syncthreads();
  }
#pragma unroll
  for (int nt = 0; nt < 2; ++nt) {
    const int nl = nh + nt * 16 + lm;
    const float bv = bh[nb + nl];
#pragma unroll
    for (int j = 0; j < 8; ++j) {
      const int ml = mq + j + 8 * halfsel;
      H1mix[(size_t)(mb + ml) * HID + nb + nl] = (__bf16)sigf(acc[nt][j] + bv);
    }
  }
}

// --------------------------- persistent scan kernel ------------------------
// grid = (HID/64, B_SZ/64) = 32 blocks, all co-resident on 256 WGPs.
__global__ __launch_bounds__(256) void k_scan(
    const __bf16* __restrict__ Xall,
    const __bf16* __restrict__ Wt0, const __bf16* __restrict__ Wt1,
    const __bf16* __restrict__ Wtm,
    const float* __restrict__ biasA0, const float* __restrict__ biasA1,
    const float* __restrict__ bh,
    float* __restrict__ Cw, float* __restrict__ C1w,
    __bf16* __restrict__ Hb0, __bf16* __restrict__ Hb1,
    __bf16* __restrict__ H1b0, __bf16* __restrict__ H1b1,
    __bf16* __restrict__ H1m,
    unsigned* __restrict__ ctr) {
  __shared__ __align__(16) unsigned char smem[4 * 64 * 64 * 4];     // 64 KB
  __bf16* As = (__bf16*)smem;                         // [2][64*LDP]   10 KB
  __bf16* Bs = (__bf16*)(smem + 2 * 64 * LDP * 2);    // [2][4*64*LDP] 40 KB
  float*  Gact = (float*)smem;                        // [4][64][64] (aliased)

  const int nb = blockIdx.x * 64;
  const int mb = blockIdx.y * 64;
  const unsigned nblk = gridDim.x * gridDim.y;
  unsigned bar = 0;

  __bf16 *Hc = Hb0, *Hn = Hb1, *H1c = H1b0, *H1n = H1b1;
#pragma unroll 1
  for (int tt = 0; tt < T_SZ; ++tt) {
    const __bf16* Xt = Xall + (size_t)tt * B_SZ * EMBD;
    cell_tile(Xt, Hc, Wt0, biasA0, Cw, Hn, mb, nb, As, Bs, Gact);
    grid_barrier(ctr, ++bar * nblk);
    mix_tile(Hn, H1c, Wtm, bh, H1m, mb, nb, As, Bs);
    grid_barrier(ctr, ++bar * nblk);
    cell_tile(Xt, H1m, Wt1, biasA1, C1w, H1n, mb, nb, As, Bs, Gact);
    grid_barrier(ctr, ++bar * nblk);
    __bf16* tp = Hc; Hc = Hn; Hn = tp;
    tp = H1c; H1c = H1n; H1n = tp;
  }
}

// --------------------------- final projection ------------------------------
// out[b][n] = H1[b][:] . Wtq[n][:] + bq[n]   (K = 512, N = 32000, f32 out)
__global__ __launch_bounds__(256) void k_proj(
    const __bf16* __restrict__ H1, const __bf16* __restrict__ Wtq, // [NCLASS][HID]
    const float* __restrict__ bq, float* __restrict__ out) {
  __shared__ __align__(16) __bf16 As[2 * 64 * LDP];
  __shared__ __align__(16) __bf16 Bs[2 * 64 * LDP];
  const int t = threadIdx.x;
  const int wave = t >> 5, lane = t & 31;
  const int mq = (wave & 3) * 16;
  const int nh = (wave >> 2) * 32;
  const int halfsel = lane >> 4, lm = lane & 15;
  const int nb = blockIdx.x * 64, mb = blockIdx.y * 64;
  const int arow = t >> 2, acol = (t & 3) * 8;
  constexpr int NC = HID / 32;         // 16 chunks
  constexpr int SLOT = 64 * LDP;

  f32x8 acc[2] = {};

  auto issue = [&](int kc, int s) {
    const int ks = kc * 32;
    cp16(As + s * SLOT + arow * LDP + acol,
         H1 + (size_t)(mb + arow) * HID + ks + acol);
    cp16(Bs + s * SLOT + arow * LDP + acol,
         Wtq + (size_t)(nb + arow) * HID + ks + acol);
  };

  issue(0, 0);
  for (int kc = 0; kc < NC; ++kc) {
    const int s = kc & 1;
    if (kc + 1 < NC) { issue(kc + 1, s ^ 1); stage_wait<2>(); }
    else             { stage_wait<0>(); }
    __syncthreads();
    const __bf16* Asl = As + s * SLOT;
    const __bf16* Bsl = Bs + s * SLOT;
    const bf16x16 af = frag_ld(&Asl[(mq + lm) * LDP + halfsel * 8], 16);
#pragma unroll
    for (int nt = 0; nt < 2; ++nt) {
      const bf16x16 bfr = frag_ld(&Bsl[(nh + nt * 16 + lm) * LDP + halfsel * 8], 16);
      acc[nt] = __builtin_amdgcn_wmma_f32_16x16x32_bf16(
          false, af, false, bfr, (short)0, acc[nt], false, false);
    }
    __syncthreads();
  }
#pragma unroll
  for (int nt = 0; nt < 2; ++nt) {
    const int nl = nh + nt * 16 + lm;
    const float bv = bq[nb + nl];
#pragma unroll
    for (int j = 0; j < 8; ++j) {
      const int ml = mq + j + 8 * halfsel;
      out[(size_t)(mb + ml) * NCLASS + nb + nl] = acc[nt][j] + bv;
    }
  }
}

// --------------------------- host driver -----------------------------------

extern "C" void kernel_launch(void* const* d_in, const int* in_sizes, int n_in,
                              void* d_out, int out_size, void* d_ws, size_t ws_size,
                              hipStream_t stream) {
  (void)in_sizes; (void)n_in; (void)out_size; (void)ws_size;
  const int*   X = (const int*)d_in[0];
  const float* H = (const float*)d_in[1];
  const float* C = (const float*)d_in[2];
  const float* E = (const float*)d_in[3];
  const float *Wx0[4], *Wh0[4], *b0[4], *Wx1[4], *Wh1[4], *b1[4];
  for (int g = 0; g < 4; ++g) {            // gate order i,f,o,c per dict order
    const int base = 4 + g * 6;
    Wx0[g] = (const float*)d_in[base + 0];
    Wh0[g] = (const float*)d_in[base + 1];
    b0[g]  = (const float*)d_in[base + 2];
    Wx1[g] = (const float*)d_in[base + 3];
    Wh1[g] = (const float*)d_in[base + 4];
    b1[g]  = (const float*)d_in[base + 5];
  }
  const float* W_xh = (const float*)d_in[28];
  const float* W_hh = (const float*)d_in[29];
  const float* b_h  = (const float*)d_in[30];
  const float* W_hq = (const float*)d_in[31];
  const float* b_q  = (const float*)d_in[32];

  char* ws = (char*)d_ws;
  size_t off = 0;
  auto carve = [&](size_t bytes) -> char* {
    char* p = ws + off;
    off = (off + bytes + 255) & ~(size_t)255;
    return p;
  };
  __bf16*   Xall = (__bf16*)carve((size_t)T_SZ * B_SZ * EMBD * 2);
  __bf16*   Wt0  = (__bf16*)carve((size_t)4 * HID * KTOT * 2);
  __bf16*   Wt1  = (__bf16*)carve((size_t)4 * HID * KTOT * 2);
  __bf16*   Wtm  = (__bf16*)carve((size_t)HID * 2 * HID * 2);
  __bf16*   Wtq  = (__bf16*)carve((size_t)NCLASS * HID * 2);
  __bf16*   Hb0  = (__bf16*)carve((size_t)B_SZ * HID * 2);
  __bf16*   Hb1  = (__bf16*)carve((size_t)B_SZ * HID * 2);
  __bf16*   H1b0 = (__bf16*)carve((size_t)B_SZ * HID * 2);
  __bf16*   H1b1 = (__bf16*)carve((size_t)B_SZ * HID * 2);
  __bf16*   H1m  = (__bf16*)carve((size_t)B_SZ * HID * 2);
  float*    Cw   = (float*)carve((size_t)B_SZ * HID * 4);
  float*    C1w  = (float*)carve((size_t)B_SZ * HID * 4);
  float*    bA0  = (float*)carve((size_t)4 * HID * 4);
  float*    bA1  = (float*)carve((size_t)4 * HID * 4);
  unsigned* ctr  = (unsigned*)carve(256);

  // per-call prep: weight transpose/convert, bias pack, gather, state init
  for (int g = 0; g < 4; ++g) {
    k_cat_tr<<<1536, 256, 0, stream>>>(Wx0[g], EMBD, Wh0[g], HID, HID,
                                       Wt0 + (size_t)g * HID * KTOT);
    k_cat_tr<<<1536, 256, 0, stream>>>(Wx1[g], EMBD, Wh1[g], HID, HID,
                                       Wt1 + (size_t)g * HID * KTOT);
  }
  k_cat_tr<<<2048, 256, 0, stream>>>(W_xh, HID, W_hh, HID, HID, Wtm);
  k_cat_tr<<<4096, 256, 0, stream>>>(W_hq, HID, nullptr, 0, NCLASS, Wtq);
  k_pack_bias<<<8, 256, 0, stream>>>(b0[0], b0[1], b0[2], b0[3], bA0);
  k_pack_bias<<<8, 256, 0, stream>>>(b1[0], b1[1], b1[2], b1[3], bA1);
  k_gather<<<8192, 256, 0, stream>>>(X, E, Xall);
  k_init<<<512, 256, 0, stream>>>(H, C, Hb0, H1b0, Cw, C1w, ctr);

  // whole scan in one persistent kernel (32 blocks, 768 internal barriers)
  k_scan<<<dim3(HID / 64, B_SZ / 64), 256, 0, stream>>>(
      Xall, Wt0, Wt1, Wtm, bA0, bA1, b_h, Cw, C1w,
      Hb0, Hb1, H1b0, H1b1, H1m, ctr);

  // T is even, so the final H1 lives in H1b0
  k_proj<<<dim3(NCLASS / 64, B_SZ / 64), 256, 0, stream>>>(H1b0, Wtq, b_q,
                                                           (float*)d_out);
}
